// Model_25778393711248
// MI455X (gfx1250) — matrix operand order repaired and checked by reading, not compiled, not run on hardware
//
#include <hip/hip_runtime.h>
#include <hip/hip_bf16.h>

typedef __attribute__((ext_vector_type(16))) _Float16 v16h;
typedef __attribute__((ext_vector_type(8)))  _Float16 v8h;
typedef __attribute__((ext_vector_type(8)))  float    v8f;

#define SEQ   128
#define BATCH 32
#define HID   512
#define GATES 2048
#define EMB   300
#define EMBP  320
#define VOCAB 32000
#define MTOT  8192          // 2 * BATCH * SEQ rows of logits
#define LDSP  520           // padded LDS row (halfs) to spread banks
#define NPANEL 64           // lin_w columns resident per block in K4
#define JBLK  32            // hid-tiles (blocks) per direction in K3

// ---------------- WMMA helpers (16x16x32 f16; A and B both use contiguous k = 16h + s) ----

__device__ __forceinline__ v8f wmma_f16(v16h a, v16h b, v8f c) {
    return __builtin_amdgcn_wmma_f32_16x16x32_f16(
        /*neg_a=*/false, a, /*neg_b=*/false, b,
        /*c_mod=*/(short)0, c, /*reuse_a=*/false, /*reuse_b=*/false);
}

// A fragment: A row-major [M][ld] halfs; tile origin (m0, k0).
__device__ __forceinline__ v16h load_a16(const _Float16* A, int ld, int m0, int k0, int lane) {
    const int row  = m0 + (lane & 15);
    const int koff = k0 + ((lane >> 4) << 4);
    const _Float16* p = A + row * ld + koff;
    v8h lo = *(const v8h*)p;
    v8h hi = *(const v8h*)(p + 8);
    v16h a;
#pragma unroll
    for (int j = 0; j < 8; ++j) { a[j] = lo[j]; a[8 + j] = hi[j]; }
    return a;
}

// B fragment: B stored [N][ld] row-major, tile (n0, k0).
__device__ __forceinline__ v16h load_b16(const _Float16* B, int ld, int n0, int k0, int lane) {
    const int col  = n0 + (lane & 15);
    const int koff = k0 + ((lane >> 4) << 4);
    return *(const v16h*)(B + col * ld + koff);
}

__device__ __forceinline__ float sigmf(float x) { return 1.f / (1.f + __expf(-x)); }
__device__ __forceinline__ float clip3(float x) { return fminf(fmaxf(x, -3.f), 3.f); }

// CDNA5 async global->LDS copy of one 16B chunk (ASYNCcnt-tracked).
__device__ __forceinline__ void async_copy16(_Float16* lds_dst, const _Float16* gsrc) {
    const unsigned la = (unsigned)(uintptr_t)lds_dst;            // low 32b = LDS offset
    const unsigned long long ga = (unsigned long long)(uintptr_t)gsrc;
    asm volatile("global_load_async_to_lds_b128 %0, %1, off" :: "v"(la), "v"(ga) : "memory");
}
__device__ __forceinline__ void async_wait0() {
    asm volatile("s_wait_asynccnt 0x0" ::: "memory");
}

// Monotonic-counter device-scope barrier across a group of `members` blocks.
__device__ __forceinline__ void gbar(unsigned* ctr, unsigned target) {
    __threadfence();
    __syncthreads();
    if (threadIdx.x == 0) {
        __hip_atomic_fetch_add(ctr, 1u, __ATOMIC_RELEASE, __HIP_MEMORY_SCOPE_AGENT);
        while (__hip_atomic_load(ctr, __ATOMIC_ACQUIRE, __HIP_MEMORY_SCOPE_AGENT) < target)
            __builtin_amdgcn_s_sleep(1);
    }
    __syncthreads();
}

// ---------------- K-1: per-call init (counters + h state) ------------------

__global__ void init_kernel(unsigned* ctrs, _Float16* h_glob, int nh) {
    const int i = blockIdx.x * blockDim.x + threadIdx.x;
    if (i < 2) ctrs[i] = 0u;
    for (int k = i; k < nh; k += gridDim.x * blockDim.x) h_glob[k] = (_Float16)0.f;
}

// ---------------- K0: f32 -> f16 weight conversion with K padding ----------

__global__ void cvt_f16_kernel(const float* __restrict__ src, _Float16* __restrict__ dst,
                               int rows, int scols, int dcols) {
    const int total = rows * dcols;
    for (int idx = blockIdx.x * blockDim.x + threadIdx.x; idx < total;
         idx += gridDim.x * blockDim.x) {
        const int r = idx / dcols, c = idx - r * dcols;
        dst[idx] = (c < scols) ? (_Float16)src[r * scols + c] : (_Float16)0.f;
    }
}

// ---------------- K1: embedding gather -> x_f16 [4096][320] ----------------

__global__ void embed_kernel(const int* __restrict__ tokens, const float* __restrict__ emb,
                             _Float16* __restrict__ x) {
    const int m = blockIdx.x;                 // m = b*SEQ + t
    const int tok = tokens[m];
    for (int k = threadIdx.x; k < EMBP; k += blockDim.x)
        x[m * EMBP + k] = (k < EMB) ? (_Float16)emb[tok * EMB + k] : (_Float16)0.f;
}

// ---------------- K2: gates_pre = x @ w_ih^T + b  -> [dir][t][b][2048] -----

__global__ __launch_bounds__(128)
void input_gemm_kernel(const _Float16* __restrict__ x,     // [4096][320]
                       const _Float16* __restrict__ w_ih,  // [2][2048][320]
                       const float* __restrict__ fwd_b, const float* __restrict__ bwd_b,
                       float* __restrict__ gates_pre) {    // [2][128][32][2048]
    const int dir  = blockIdx.z;
    const int lane = threadIdx.x & 31;
    const int wave = threadIdx.x >> 5;
    const int m0   = blockIdx.x * 16;
    const int nt0  = blockIdx.y * 16 + wave * 4;
    const _Float16* Bm  = w_ih + (size_t)dir * GATES * EMBP;
    const float* bias   = dir ? bwd_b : fwd_b;

    v8f acc[4];
#pragma unroll
    for (int q = 0; q < 4; ++q) {
        const float bv = bias[(nt0 + q) * 16 + (lane & 15)];
#pragma unroll
        for (int v = 0; v < 8; ++v) acc[q][v] = bv;
    }
    for (int kc = 0; kc < EMBP; kc += 32) {
        const v16h a = load_a16(x, EMBP, m0, kc, lane);
#pragma unroll
        for (int q = 0; q < 4; ++q) {
            const v16h b = load_b16(Bm, EMBP, (nt0 + q) * 16, kc, lane);
            acc[q] = wmma_f16(a, b, acc[q]);
        }
    }
    const int rowhi = (lane >> 4) << 3;
#pragma unroll
    for (int q = 0; q < 4; ++q) {
        const int n = (nt0 + q) * 16 + (lane & 15);
#pragma unroll
        for (int v = 0; v < 8; ++v) {
            const int m = m0 + v + rowhi;
            const int bi = m >> 7, t = m & 127;
            gates_pre[(((size_t)(dir * SEQ + t) * BATCH + bi) * GATES) + n] = acc[q][v];
        }
    }
}

// ---------------- K3: LSTM recurrence, weight-resident across 32 blocks ----
// grid (32, 2): blockIdx.x = hid-tile j (16 cols), blockIdx.y = dir.
// Each block keeps its 64-row w_hh slice + 16-row w_proj slice in LDS for all
// 128 steps; h / h_pre (32x512 f16) are exchanged via L2 with spin barriers.

__global__ __launch_bounds__(64)
void lstm_recurrent_kernel(const float* __restrict__ gp_all,  // [2][128][32][2048]
                           const _Float16* __restrict__ w_hh, // [2][2048][512]
                           const _Float16* __restrict__ w_proj,// [2][512][512]
                           _Float16* __restrict__ h_glob,     // [2][32][512]
                           _Float16* __restrict__ hp_glob,    // [2][32][512]
                           unsigned* __restrict__ ctrs,       // [2]
                           _Float16* __restrict__ h_all) {    // [8192][512]
    __shared__ _Float16 sWhh[64][LDSP];   // rows: g*16 + col
    __shared__ _Float16 sWpr[16][LDSP];
    const int j    = blockIdx.x;
    const int dir  = blockIdx.y;
    const int lane = threadIdx.x & 31;
    const int wave = threadIdx.x >> 5;    // 2 waves: one 16-row batch tile each
    const int m0   = wave * 16;
    const int rowhi = (lane >> 4) << 3;
    const _Float16* whh = w_hh   + (size_t)dir * GATES * HID;
    const _Float16* wpr = w_proj + (size_t)dir * HID * HID;
    _Float16* hd  = h_glob  + (size_t)dir * BATCH * HID;
    _Float16* hpd = hp_glob + (size_t)dir * BATCH * HID;
    unsigned* ctr = ctrs + dir;

    // stage weight slices into LDS with CDNA5 async copies (once)
    for (int c = threadIdx.x; c < 64 * (HID / 8); c += blockDim.x) {
        const int r = c >> 6, cc = (c & 63) << 3;
        const int g = r >> 4, col = r & 15;
        async_copy16(&sWhh[r][cc], whh + (size_t)(g * HID + j * 16 + col) * HID + cc);
    }
    for (int c = threadIdx.x; c < 16 * (HID / 8); c += blockDim.x) {
        const int r = c >> 6, cc = (c & 63) << 3;
        async_copy16(&sWpr[r][cc], wpr + (size_t)(j * 16 + r) * HID + cc);
    }
    async_wait0();
    __syncthreads();

    v8f cstate;
#pragma unroll
    for (int v = 0; v < 8; ++v) cstate[v] = 0.f;

    unsigned epoch = 0;
    for (int step = 0; step < SEQ; ++step) {
        const int tt = dir ? (SEQ - 1 - step) : step;
        const float* gp = gp_all + (size_t)(dir * SEQ + tt) * (BATCH * GATES);

        // gates slice: [16 batch x 16 cols] x 4 gates = gp + h @ w_hh_slice^T
        v8f acc[4];
#pragma unroll
        for (int g = 0; g < 4; ++g) {
            const int n = g * HID + j * 16 + (lane & 15);
#pragma unroll
            for (int v = 0; v < 8; ++v)
                acc[g][v] = gp[(m0 + v + rowhi) * GATES + n];
        }
        for (int kc = 0; kc < HID; kc += 32) {
            const v16h a = load_a16(hd, HID, m0, kc, lane);
#pragma unroll
            for (int g = 0; g < 4; ++g) {
                const v16h b = load_b16(&sWhh[0][0], LDSP, g * 16, kc, lane);
                acc[g] = wmma_f16(a, b, acc[g]);
            }
        }
#pragma unroll
        for (int v = 0; v < 8; ++v) {
            const float ig = sigmf(acc[0][v]);
            const float fg = sigmf(acc[1][v]);
            const float gg = tanhf(acc[2][v]);
            const float og = sigmf(acc[3][v]);
            const float c = clip3(fg * cstate[v] + ig * gg);
            cstate[v] = c;
            hpd[(m0 + v + rowhi) * HID + j * 16 + (lane & 15)] =
                (_Float16)(og * tanhf(c));
        }
        gbar(ctr, JBLK * (++epoch));      // all h_pre published

        // projection slice: h[:, j*16..+15] = clip(h_pre @ w_proj_slice^T)
        v8f pacc;
#pragma unroll
        for (int v = 0; v < 8; ++v) pacc[v] = 0.f;
        for (int kc = 0; kc < HID; kc += 32) {
            const v16h a = load_a16(hpd, HID, m0, kc, lane);
            const v16h b = load_b16(&sWpr[0][0], LDSP, 0, kc, lane);
            pacc = wmma_f16(a, b, pacc);
        }
#pragma unroll
        for (int v = 0; v < 8; ++v) {
            const float h = clip3(pacc[v]);
            const int bi = m0 + v + rowhi;
            const int col = j * 16 + (lane & 15);
            hd[bi * HID + col] = (_Float16)h;
            h_all[((size_t)(dir * (BATCH * SEQ) + bi * SEQ + tt)) * HID + col] =
                (_Float16)h;
        }
        gbar(ctr, JBLK * (++epoch));      // all h published
    }
}

// ---------------- K4: logits = h_all @ lin_w^T + lin_b  (dominant GEMM) ----
// Each block owns a 64-column lin_w panel resident in LDS (read from HBM once)
// and sweeps all 512 m-tiles; A streams from L2.

__global__ __launch_bounds__(256)
void output_gemm_kernel(const _Float16* __restrict__ h_all, // [8192][512]
                        const _Float16* __restrict__ lw,    // [32000][512]
                        const float* __restrict__ lb,
                        float* __restrict__ out) {          // [8192][32000]
    __shared__ _Float16 shB[NPANEL][LDSP];
    const int lane = threadIdx.x & 31;
    const int wave = threadIdx.x >> 5;
    const int n0   = blockIdx.x * NPANEL;

    // async-stage the 64x512 lin_w panel
    for (int c = threadIdx.x; c < NPANEL * (HID / 8); c += blockDim.x) {
        const int r = c >> 6, cc = (c & 63) << 3;
        async_copy16(&shB[r][cc], lw + (size_t)(n0 + r) * HID + cc);
    }
    async_wait0();
    __syncthreads();

    float bias[4];
#pragma unroll
    for (int q = 0; q < 4; ++q) bias[q] = lb[n0 + q * 16 + (lane & 15)];

    const int rowhi = (lane >> 4) << 3;
    for (int mt = wave; mt < MTOT / 16; mt += 8) {
        const int m0 = mt * 16;
        if (mt + 8 < MTOT / 16)          // prefetch next A tile (global_prefetch)
            __builtin_prefetch(h_all + (size_t)(m0 + 128) * HID + (threadIdx.x & 63) * 16, 0, 0);
        v8f acc[4];
#pragma unroll
        for (int q = 0; q < 4; ++q)
#pragma unroll
            for (int v = 0; v < 8; ++v) acc[q][v] = bias[q];
        for (int kc = 0; kc < HID; kc += 32) {
            const v16h a = load_a16(h_all, HID, m0, kc, lane);
#pragma unroll
            for (int q = 0; q < 4; ++q) {
                const v16h b = load_b16(&shB[0][0], LDSP, q * 16, kc, lane);
                acc[q] = wmma_f16(a, b, acc[q]);
            }
        }
#pragma unroll
        for (int q = 0; q < 4; ++q) {
            const int n = n0 + q * 16 + (lane & 15);
#pragma unroll
            for (int v = 0; v < 8; ++v)
                out[(size_t)(m0 + v + rowhi) * VOCAB + n] = acc[q][v];
        }
    }
}

// ---------------- host glue ------------------------------------------------

extern "C" void kernel_launch(void* const* d_in, const int* in_sizes, int n_in,
                              void* d_out, int out_size, void* d_ws, size_t ws_size,
                              hipStream_t stream) {
    (void)in_sizes; (void)n_in; (void)out_size; (void)ws_size;
    const int*   tokens     = (const int*)d_in[0];
    const float* emb_table  = (const float*)d_in[1];
    const float* fwd_w_ih   = (const float*)d_in[2];
    const float* fwd_w_hh   = (const float*)d_in[3];
    const float* fwd_b      = (const float*)d_in[4];
    const float* fwd_w_proj = (const float*)d_in[5];
    const float* bwd_w_ih   = (const float*)d_in[6];
    const float* bwd_w_hh   = (const float*)d_in[7];
    const float* bwd_b      = (const float*)d_in[8];
    const float* bwd_w_proj = (const float*)d_in[9];
    const float* lin_w      = (const float*)d_in[10];
    const float* lin_b      = (const float*)d_in[11];
    float* out = (float*)d_out;

    char* ws = (char*)d_ws;
    auto carve = [&](size_t bytes) -> char* {
        char* p = ws; ws += (bytes + 255) & ~(size_t)255; return p;
    };
    _Float16* x_f16     = (_Float16*)carve((size_t)BATCH * SEQ * EMBP * 2);
    _Float16* w_ih_f16  = (_Float16*)carve((size_t)2 * GATES * EMBP * 2);
    _Float16* w_hh_f16  = (_Float16*)carve((size_t)2 * GATES * HID * 2);
    _Float16* w_pr_f16  = (_Float16*)carve((size_t)2 * HID * HID * 2);
    _Float16* lin_w_f16 = (_Float16*)carve((size_t)VOCAB * HID * 2);
    float*    gates_pre = (float*)   carve((size_t)2 * SEQ * BATCH * GATES * 4);
    _Float16* h_all     = (_Float16*)carve((size_t)MTOT * HID * 2);
    _Float16* h_glob    = (_Float16*)carve((size_t)2 * BATCH * HID * 2);
    _Float16* hp_glob   = (_Float16*)carve((size_t)2 * BATCH * HID * 2);
    unsigned* ctrs      = (unsigned*)carve(256);

    // per-call state init (graph-replay safe)
    init_kernel<<<64, 256, 0, stream>>>(ctrs, h_glob, 2 * BATCH * HID);

    auto cvt = [&](const float* s, _Float16* d, int r, int sc, int dc) {
        int total = r * dc;
        int blocks = (total + 255) / 256; if (blocks > 4096) blocks = 4096;
        cvt_f16_kernel<<<blocks, 256, 0, stream>>>(s, d, r, sc, dc);
    };
    cvt(fwd_w_ih,   w_ih_f16,                       GATES, EMB, EMBP);
    cvt(bwd_w_ih,   w_ih_f16 + (size_t)GATES*EMBP,  GATES, EMB, EMBP);
    cvt(fwd_w_hh,   w_hh_f16,                       GATES, HID, HID);
    cvt(bwd_w_hh,   w_hh_f16 + (size_t)GATES*HID,   GATES, HID, HID);
    cvt(fwd_w_proj, w_pr_f16,                       HID,   HID, HID);
    cvt(bwd_w_proj, w_pr_f16 + (size_t)HID*HID,     HID,   HID, HID);
    cvt(lin_w,      lin_w_f16,                      VOCAB, HID, HID);

    embed_kernel<<<BATCH * SEQ, EMBP, 0, stream>>>(tokens, emb_table, x_f16);

    input_gemm_kernel<<<dim3(BATCH * SEQ / 16, GATES / (16 * 16), 2), 128, 0, stream>>>(
        x_f16, w_ih_f16, fwd_b, bwd_b, gates_pre);

    lstm_recurrent_kernel<<<dim3(JBLK, 2), 64, 0, stream>>>(
        gates_pre, w_hh_f16, w_pr_f16, h_glob, hp_glob, ctrs, h_all);

    output_gemm_kernel<<<VOCAB / NPANEL, 256, 0, stream>>>(
        h_all, lin_w_f16, lin_b, out);
}
